// GRU_78512002171284
// MI455X (gfx1250) — matrix operand order repaired and checked by reading, not compiled, not run on hardware
//
#include <hip/hip_runtime.h>

// ---------------------------------------------------------------------------
// Types
// ---------------------------------------------------------------------------
typedef __attribute__((ext_vector_type(16))) _Float16 v16h;
typedef __attribute__((ext_vector_type(8)))  float    v8f;

__device__ __forceinline__ unsigned short f2h(float f) {
    union { _Float16 h; unsigned short u; } u;
    u.h = (_Float16)f;
    return u.u;
}

__device__ __forceinline__ v16h as_v16h(uint4 lo, uint4 hi) {
    union { uint4 q[2]; v16h v; } u;
    u.q[0] = lo; u.q[1] = hi;
    return u.v;
}

#define BDIM 128   // 4 waves per block
#define NBLK 32    // persistent blocks (grid barrier participants)
#define KC   256   // K-chunk staged in LDS; divides din and H for all layers
#define BATCH 32
#define TLEN 2048

// ---------------------------------------------------------------------------
// Input transpose + f32->f16: x[B, C, T] -> xin[B, T, C]
// ---------------------------------------------------------------------------
__global__ void transpose_in(const float* __restrict__ x,
                             unsigned short* __restrict__ xin,
                             int C, int T) {
    size_t n = (size_t)BATCH * C * T;
    for (size_t i = (size_t)blockIdx.x * blockDim.x + threadIdx.x; i < n;
         i += (size_t)gridDim.x * blockDim.x) {
        size_t b = i / ((size_t)T * C);
        size_t r = i % ((size_t)T * C);
        size_t t = r / C, c = r % C;
        xin[i] = f2h(x[(b * C + c) * T + t]);
    }
}

// ---------------------------------------------------------------------------
// Weight pack: build fragment-major f16 weights for the fused [K, 4H] GEMM.
// Gate groups: 0=r (full K), 1=z (full K), 2=n_x (k<din), 3=n_h (k>=din).
// Fragment (f): 512 f16 values laid out so lane L reads its 16 contiguous
// values at f*512 + L*16.  B frag (32x16, KxN), contiguous k order chosen here:
//   lanes 0-15 : column n = L,    K = ktg*32 + 0..15
//   lanes 16-31: column n = L-16, K = ktg*32 + 16..31
// ---------------------------------------------------------------------------
__global__ void pack_w(const float* __restrict__ wih,
                       const float* __restrict__ whh,
                       unsigned short* __restrict__ wp, int din, int H) {
    const int Jt = H >> 4, Kt = (din + H) >> 5, Ktx = din >> 5, Kth = H >> 5;
    const int JtKt = Jt * Kt, base2 = 2 * JtKt, base3 = base2 + Jt * Ktx;
    const size_t nelem = ((size_t)(base3 + Jt * Kth)) << 9;
    for (size_t idx = (size_t)blockIdx.x * blockDim.x + threadIdx.x; idx < nelem;
         idx += (size_t)gridDim.x * blockDim.x) {
        int f    = (int)(idx >> 9);
        int r    = (int)(idx & 511);
        int lane = r >> 4, i = r & 15;
        int g, jt, ktg;
        if (f < JtKt)       { g = 0; jt = f / Kt;  ktg = f % Kt; }
        else if (f < base2) { int f1 = f - JtKt;  g = 1; jt = f1 / Kt;  ktg = f1 % Kt; }
        else if (f < base3) { int f2 = f - base2; g = 2; jt = f2 / Ktx; ktg = f2 % Ktx; }
        else                { int f3 = f - base3; g = 3; jt = f3 / Kth; ktg = Ktx + f3 % Kth; }
        int n    = jt * 16 + (lane & 15);
        int k    = ktg * 32 + ((lane >= 16) ? 16 : 0) + i;
        int col3 = ((g >= 2) ? 2 : g) * H + n;   // row in [3H, din|H] weight
        float v  = (k < din) ? wih[(size_t)col3 * din + k]
                             : whh[(size_t)col3 * H + (k - din)];
        wp[idx] = f2h(v);
    }
}

// bias[0:H]=b_ih_r+b_hh_r, [H:2H]=b_ih_z+b_hh_z, [2H:3H]=b_ih_n, [3H:4H]=b_hh_n
__global__ void pack_bias(const float* __restrict__ bih,
                          const float* __restrict__ bhh,
                          float* __restrict__ bo, int H) {
    int j = blockIdx.x * blockDim.x + threadIdx.x;
    if (j < H) {
        bo[j]         = bih[j]         + bhh[j];
        bo[H + j]     = bih[H + j]     + bhh[H + j];
        bo[2 * H + j] = bih[2 * H + j];
        bo[3 * H + j] = bhh[2 * H + j];
    }
}

__global__ void init_state(float* hf32, unsigned short* hf16,
                           unsigned* bar, int n) {
    for (int i = blockIdx.x * blockDim.x + threadIdx.x; i < n;
         i += gridDim.x * blockDim.x) {
        hf32[i] = 0.f;
        hf16[i] = 0;
    }
    if (blockIdx.x == 0 && threadIdx.x < 8) bar[threadIdx.x] = 0;
}

// ---------------------------------------------------------------------------
// Persistent GRU recurrence.  Per timestep:
//   A = [x_t ; h]  (32 x Ktot, f16, staged into LDS with CDNA5 async copies;
//                   every KC-chunk is uniformly x- or h-sourced since KC|din)
//   out[32, 4H] = A @ W'   via v_wmma_f32_16x16x32_f16 (B frags double-buffered)
//   gate combine + h update in registers, then device-wide barrier.
// Wave w owns item (mt = w&1, jt = w>>1): rows mt*16..+15, cols jt*16..+15.
// ---------------------------------------------------------------------------
__global__ void __launch_bounds__(BDIM, 1)
gru_recur(const unsigned short* __restrict__ xin,  // [B, T, din] f16 bits
          unsigned short* __restrict__ yout,       // [B, T, H]  f16 bits (or null)
          float* __restrict__ fout,                // [B, H, T]  f32      (or null)
          const unsigned short* __restrict__ wpack,
          const float* __restrict__ bias,          // [4H]
          float* __restrict__ hf32,                // [32, H]
          unsigned short* __restrict__ hf16,       // [32, H]
          unsigned* __restrict__ bar,
          int din, int H, int T) {
    const int Ktot = din + H;
    const int Kt = Ktot >> 5, Ktx = din >> 5, Kth = H >> 5, Jt = H >> 4;
    const int JtKt = Jt * Kt, base2 = 2 * JtKt, base3 = base2 + Jt * Ktx;
    const int nch = Ktot / KC;

    const int tid  = threadIdx.x;
    const int lane = tid & 31;
    const int wid  = tid >> 5;
    const int gw   = blockIdx.x * (BDIM / 32) + wid;   // 0..127
    const int items = 2 * Jt;                          // <= 128
    const bool active = gw < items;
    const int mt = gw & 1;
    const int jt = gw >> 1;

    const int LROW = KC + 8;                           // pad rows vs bank conflicts
    __shared__ unsigned short As[BATCH * (KC + 8)];
    // LDS byte offset of As[0] (generic LDS pointer: low 32 bits = LDS offset)
    const unsigned lds_base = (unsigned)(uintptr_t)&As[0];

    // combine-phase per-lane constants (C-layout: lane -> column, vgpr -> row)
    const int jc = jt * 16 + (lane & 15);
    float br = 0.f, bz = 0.f, bnx = 0.f, bnh = 0.f;
    if (active) {
        br  = bias[jc];
        bz  = bias[H + jc];
        bnx = bias[2 * H + jc];
        bnh = bias[3 * H + jc];
    }
    const int mrow0 = mt * 16 + ((lane >> 4) << 3);    // +8 for lanes 16..31
    const int arow  = mt * 16 + (lane & 15);           // A-frag row for this lane
    const int khalf = (lane >> 4) ? 16 : 0;            // A-frag K sub-offset

    // Per-thread staging coordinates (fixed across chunks/timesteps):
    // thread covers vectors v = tid + 128*i, i=0..7 ; row = v>>5, c8 = v&31
    const uint4* wq = (const uint4*)wpack + (size_t)lane * 2;

    for (int t = 0; t < T; ++t) {
        v8f aR  = {0.f, 0.f, 0.f, 0.f, 0.f, 0.f, 0.f, 0.f};
        v8f aZ  = aR, aCX = aR, aDH = aR;

        for (int ch = 0; ch < nch; ++ch) {
            const int kc0 = ch * KC;
            // ---- async stage A chunk [32, KC] global -> LDS --------------
            // KC divides din, so the whole chunk comes from one source:
            // uniform scalar select of base pointer + row stride (no lane
            // divergence, no exec waterfall).
            const bool isx = kc0 < din;
            const unsigned short* base =
                isx ? (xin + (size_t)t * din + kc0) : (hf16 + (kc0 - din));
            const size_t rstride = isx ? (size_t)T * din : (size_t)H;
            for (int v = tid; v < BATCH * (KC / 8); v += BDIM) {
                int row = v >> 5;            // KC/8 == 32 vec8 per row
                int c8  = v & 31;
                const unsigned short* gp = base + (size_t)row * rstride + c8 * 8;
                unsigned lds_off =
                    lds_base + (unsigned)((row * LROW + c8 * 8) * 2);
                asm volatile("global_load_async_to_lds_b128 %0, %1, off"
                             :: "v"(lds_off),
                                "v"((unsigned long long)(uintptr_t)gp)
                             : "memory");
                // warm next timestep's x slice (speculative, dropped on fault)
                if (isx) __builtin_prefetch(gp + din, 0, 1);
            }
            asm volatile("s_wait_asynccnt 0" ::: "memory");
            __syncthreads();

            if (active) {
                const int ktg0 = kc0 >> 5;
                uint4 bb[2][6];
                auto loadB = [&](int kl, int buf) {
                    int ktg = ktg0 + kl;
                    int fR  = jt * Kt + ktg;
                    int fZ  = JtKt + fR;
                    int fN  = (ktg < Ktx) ? (base2 + jt * Ktx + ktg)
                                          : (base3 + jt * Kth + (ktg - Ktx));
                    bb[buf][0] = wq[(size_t)fR * 64];
                    bb[buf][1] = wq[(size_t)fR * 64 + 1];
                    bb[buf][2] = wq[(size_t)fZ * 64];
                    bb[buf][3] = wq[(size_t)fZ * 64 + 1];
                    bb[buf][4] = wq[(size_t)fN * 64];
                    bb[buf][5] = wq[(size_t)fN * 64 + 1];
                };
                loadB(0, 0);
#pragma unroll
                for (int kl = 0; kl < KC / 32; ++kl) {
                    const int buf = kl & 1;
                    if (kl + 1 < KC / 32) loadB(kl + 1, buf ^ 1);

                    // A fragment, same contiguous k order as B -> lane holds
                    // K = base+{0..15}, base = ktg*32 + khalf
                    const uint4* ap =
                        (const uint4*)&As[arow * LROW + kl * 32 + khalf];
                    v16h av = as_v16h(ap[0], ap[1]);

                    aR = __builtin_amdgcn_wmma_f32_16x16x32_f16(
                        false, av, false, as_v16h(bb[buf][0], bb[buf][1]),
                        (short)0, aR, false, false);
                    aZ = __builtin_amdgcn_wmma_f32_16x16x32_f16(
                        false, av, false, as_v16h(bb[buf][2], bb[buf][3]),
                        (short)0, aZ, false, false);
                    const int ktg = ktg0 + kl;
                    if (ktg < Ktx)
                        aCX = __builtin_amdgcn_wmma_f32_16x16x32_f16(
                            false, av, false, as_v16h(bb[buf][4], bb[buf][5]),
                            (short)0, aCX, false, false);
                    else
                        aDH = __builtin_amdgcn_wmma_f32_16x16x32_f16(
                            false, av, false, as_v16h(bb[buf][4], bb[buf][5]),
                            (short)0, aDH, false, false);
                }
            }
            __syncthreads();
        }

        // ---- gate combine + h update (pure VALU, register resident) ----
        if (active) {
#pragma unroll
            for (int e = 0; e < 8; ++e) {
                int m = mrow0 + e;
                float hold = hf32[m * H + jc];
                float r = 1.f / (1.f + __expf(-(aR[e] + br)));
                float z = 1.f / (1.f + __expf(-(aZ[e] + bz)));
                float n = tanhf(aCX[e] + bnx + r * (aDH[e] + bnh));
                float hn = (1.f - z) * n + z * hold;
                hf32[m * H + jc] = hn;
                hf16[m * H + jc] = f2h(hn);
                if (yout) yout[((size_t)m * T + t) * H + jc] = f2h(hn);
                if (fout) fout[((size_t)m * H + jc) * T + t] = hn;
            }
        }

        // ---- device-wide sense-reversing barrier (publish h) ----
        __threadfence();
        __syncthreads();
        if (tid == 0) {
            unsigned gen = __hip_atomic_load(&bar[1], __ATOMIC_RELAXED,
                                             __HIP_MEMORY_SCOPE_AGENT);
            unsigned arrived = __hip_atomic_fetch_add(
                &bar[0], 1u, __ATOMIC_ACQ_REL, __HIP_MEMORY_SCOPE_AGENT);
            if (arrived == (unsigned)(NBLK - 1)) {
                __hip_atomic_store(&bar[0], 0u, __ATOMIC_RELAXED,
                                   __HIP_MEMORY_SCOPE_AGENT);
                __hip_atomic_store(&bar[1], gen + 1u, __ATOMIC_RELEASE,
                                   __HIP_MEMORY_SCOPE_AGENT);
            } else {
                while (__hip_atomic_load(&bar[1], __ATOMIC_ACQUIRE,
                                         __HIP_MEMORY_SCOPE_AGENT) == gen) {
                    __builtin_amdgcn_s_sleep(2);
                }
            }
        }
        __syncthreads();
        __threadfence();
    }
}

// ---------------------------------------------------------------------------
// Host launcher
// ---------------------------------------------------------------------------
extern "C" void kernel_launch(void* const* d_in, const int* in_sizes, int n_in,
                              void* d_out, int out_size, void* d_ws,
                              size_t ws_size, hipStream_t stream) {
    (void)in_sizes; (void)n_in; (void)out_size; (void)ws_size;

    const float* x = (const float*)d_in[0];

    // ---- carve workspace (aliased across sequential layers) ----
    char* ws = (char*)d_ws;
    size_t off = 0;
    auto carve = [&](size_t bytes) -> char* {
        char* p = ws + off;
        off += (bytes + 255) & ~(size_t)255;
        return p;
    };
    unsigned short* R0 = (unsigned short*)carve((size_t)64  << 20); // xin / y01
    unsigned short* R1 = (unsigned short*)carve((size_t)128 << 20); // y00 / y10
    unsigned short* WP = (unsigned short*)carve((size_t)12  << 20); // packed W
    float*          BI = (float*)carve(4 * 1024 * sizeof(float));
    float*          HF = (float*)carve(32 * 1024 * sizeof(float));
    unsigned short* HB = (unsigned short*)carve(32 * 1024 * 2);
    unsigned*       BA = (unsigned*)carve(256);

    const int C0 = 256, T = TLEN;
    transpose_in<<<512, 256, 0, stream>>>(x, R0, C0, T);

    struct Lyr {
        int din, H, wbase;
        const unsigned short* in;
        unsigned short* out;
        float* fout;
    };
    Lyr ls[4] = {
        { 256,  512,  1, R0, R1, nullptr },
        { 512,  512,  5, R1, R0, nullptr },
        { 512, 1024,  9, R0, R1, nullptr },
        {1024, 1024, 13, R1, nullptr, (float*)d_out},
    };

    for (int l = 0; l < 4; ++l) {
        const float* wih = (const float*)d_in[ls[l].wbase + 0];
        const float* whh = (const float*)d_in[ls[l].wbase + 1];
        const float* bih = (const float*)d_in[ls[l].wbase + 2];
        const float* bhh = (const float*)d_in[ls[l].wbase + 3];
        const int din = ls[l].din, H = ls[l].H;

        pack_w<<<1024, 256, 0, stream>>>(wih, whh, WP, din, H);
        pack_bias<<<(H + 255) / 256, 256, 0, stream>>>(bih, bhh, BI, H);
        init_state<<<64, 256, 0, stream>>>(HF, HB, BA, 32 * H);
        gru_recur<<<NBLK, BDIM, 0, stream>>>(ls[l].in, ls[l].out, ls[l].fout,
                                             WP, BI, HF, HB, BA, din, H, T);
    }
}
